// BinDevianceLoss_77464030151204
// MI455X (gfx1250) — matrix operand order repaired and checked by reading, not compiled, not run on hardware
//
#include <hip/hip_runtime.h>
#include <hip/hip_bf16.h>
#include <math.h>

// ---------------------------------------------------------------------------
// BinDeviance loss, MI455X (gfx1250, wave32, WMMA).
//   X: [N=4096, D=1024] fp32 unit-norm rows, classes = contiguous groups of G=4.
//   sim = X @ X^T  computed tile-by-tile with v_wmma_f32_16x16x32_bf16 using a
//   bf16 hi/lo split (sim ~= hi*hi + hi*lo + lo*hi) for ~fp32 accuracy, fused
//   with the negative-selection epilogue so the 64MB sim matrix is never stored.
// ---------------------------------------------------------------------------

typedef __attribute__((ext_vector_type(8)))  __bf16 v8bf;
typedef __attribute__((ext_vector_type(16))) __bf16 v16bf;
typedef __attribute__((ext_vector_type(8)))  float  v8f;

#define MARGIN 0.5f
#define GCLS   4

__device__ __forceinline__ float softplusf(float x) {
    return (x > 20.f) ? x : log1pf(expf(x));
}

__device__ __forceinline__ v8f wmma_bf16(v16bf a, v16bf b, v8f c) {
    // D = A(16x32 bf16) * B(32x16 bf16) + C(16x16 f32)
    return __builtin_amdgcn_wmma_f32_16x16x32_bf16(
        /*neg_a=*/false, a, /*neg_b=*/false, b,
        /*c_mod=*/(short)0, c, /*reuse_a=*/false, /*reuse_b=*/false);
}

// A fragment (16x32, row-major source): lane ll=lane&15 is row M,
// lane half lh=lane>>4 selects K base; two 16B chunks at k0 and k0+16.
__device__ __forceinline__ v16bf load_a_frag(const __bf16* __restrict__ x,
                                             int d, int rowBase, int kk,
                                             int ll, int lh) {
    const __bf16* p = x + (size_t)(rowBase + ll) * d + kk + lh * 8;
    union { v16bf v; v8bf h[2]; } u;
    u.h[0] = *(const v8bf*)(p);
    u.h[1] = *(const v8bf*)(p + 16);
    return u.v;
}

// B fragment (32x16): column n of B == row n of X (B = X^T), so lane ll loads
// two 8-bf16 chunks of row (colBase+ll) at k = kk + 8*lh and kk + 16 + 8*lh.
__device__ __forceinline__ v16bf load_b_frag(const __bf16* __restrict__ x,
                                             int d, int colBase, int kk,
                                             int ll, int lh) {
    const __bf16* p = x + (size_t)(colBase + ll) * d + kk + lh * 8;
    union { v16bf v; v8bf h[2]; } u;
    u.h[0] = *(const v8bf*)(p);
    u.h[1] = *(const v8bf*)(p + 16);
    return u.v;
}

// ---------------- zero-init of accumulator workspace -----------------------
__global__ void init_kernel(float* __restrict__ neg_sum, int* __restrict__ sel_cnt,
                            float* __restrict__ scalars, int n) {
    int i = blockIdx.x * blockDim.x + threadIdx.x;
    if (i < n) { neg_sum[i] = 0.f; sel_cnt[i] = 0; }
    if (i < 2) scalars[i] = 0.f;   // [0]=pos_d_sum, [1]=neg_d_sum
}

// ---------------- fp32 -> bf16 hi/lo split ---------------------------------
__global__ void convert_kernel(const float* __restrict__ x,
                               __bf16* __restrict__ xhi,
                               __bf16* __restrict__ xlo, int total) {
    int i = blockIdx.x * blockDim.x + threadIdx.x;
    if (i < total) {
        float v = x[i];
        __bf16 h = (__bf16)v;
        xhi[i] = h;
        xlo[i] = (__bf16)(v - (float)h);
    }
}

// ---------------- per-row positives: min_pos, pos_loss, pos_d (fp32 exact) --
__global__ void minpos_kernel(const float* __restrict__ x,
                              float* __restrict__ min_pos,
                              float* __restrict__ pos_loss,
                              float* __restrict__ pos_d_sum,
                              int n, int d) {
    int row  = blockIdx.x * 8 + (threadIdx.x >> 5);   // one wave32 per row
    int lane = threadIdx.x & 31;
    if (row >= n) return;
    int base = (row / GCLS) * GCLS;
    const float* xr = x + (size_t)row * d;
    float a0 = 0.f, a1 = 0.f, a2 = 0.f, a3 = 0.f;
    for (int k = lane; k < d; k += 32) {
        float v = xr[k];
        a0 += v * x[(size_t)(base + 0) * d + k];
        a1 += v * x[(size_t)(base + 1) * d + k];
        a2 += v * x[(size_t)(base + 2) * d + k];
        a3 += v * x[(size_t)(base + 3) * d + k];
    }
    for (int m = 1; m < 32; m <<= 1) {
        a0 += __shfl_xor(a0, m, 32);
        a1 += __shfl_xor(a1, m, 32);
        a2 += __shfl_xor(a2, m, 32);
        a3 += __shfl_xor(a3, m, 32);
    }
    if (lane == 0) {
        float acc[GCLS] = {a0, a1, a2, a3};
        float mp = 1e30f, pl = 0.f, pd = 0.f;
        #pragma unroll
        for (int j = 0; j < GCLS; ++j) {
            if (base + j != row) {
                float s = acc[j];
                mp = fminf(mp, s);
                pl += softplusf(-2.f * (s - MARGIN));
                pd += s;
            }
        }
        min_pos[row]  = mp;
        pos_loss[row] = pl * (1.f / (float)(GCLS - 1));
        atomicAdd(pos_d_sum, pd);
    }
}

// ---------------- fused WMMA sim tile + negative epilogue ------------------
// Workgroup: 256 threads = 8 waves, tile 128(rows) x 64(cols); each wave owns
// a 32x32 sub-tile (4 accumulators). 3 WMMAs per accumulator per K-step for
// the hi/lo precision split -> 12 v_wmma per K-step per wave.
__global__ void __launch_bounds__(256)
simtile_kernel(const __bf16* __restrict__ xhi, const __bf16* __restrict__ xlo,
               const float* __restrict__ min_pos,
               float* __restrict__ neg_sum, int* __restrict__ sel_cnt,
               float* __restrict__ neg_d_sum, int n, int d) {
    const int lane    = threadIdx.x & 31;
    const int wave    = threadIdx.x >> 5;
    const int waveRow = wave & 3;                  // 0..3
    const int waveCol = wave >> 2;                 // 0..1
    const int rowBase = blockIdx.y * 128 + waveRow * 32;
    const int colBase = blockIdx.x * 64  + waveCol * 32;
    const int ll = lane & 15;
    const int lh = lane >> 4;

    v8f c00 = {}, c01 = {}, c10 = {}, c11 = {};

    for (int kk = 0; kk < d; kk += 32) {
        v16bf ah0 = load_a_frag(xhi, d, rowBase,      kk, ll, lh);
        v16bf al0 = load_a_frag(xlo, d, rowBase,      kk, ll, lh);
        v16bf ah1 = load_a_frag(xhi, d, rowBase + 16, kk, ll, lh);
        v16bf al1 = load_a_frag(xlo, d, rowBase + 16, kk, ll, lh);
        v16bf bh0 = load_b_frag(xhi, d, colBase,      kk, ll, lh);
        v16bf bl0 = load_b_frag(xlo, d, colBase,      kk, ll, lh);
        v16bf bh1 = load_b_frag(xhi, d, colBase + 16, kk, ll, lh);
        v16bf bl1 = load_b_frag(xlo, d, colBase + 16, kk, ll, lh);

        c00 = wmma_bf16(ah0, bh0, c00);
        c00 = wmma_bf16(ah0, bl0, c00);
        c00 = wmma_bf16(al0, bh0, c00);

        c01 = wmma_bf16(ah0, bh1, c01);
        c01 = wmma_bf16(ah0, bl1, c01);
        c01 = wmma_bf16(al0, bh1, c01);

        c10 = wmma_bf16(ah1, bh0, c10);
        c10 = wmma_bf16(ah1, bl0, c10);
        c10 = wmma_bf16(al1, bh0, c10);

        c11 = wmma_bf16(ah1, bh1, c11);
        c11 = wmma_bf16(ah1, bl1, c11);
        c11 = wmma_bf16(al1, bh1, c11);
    }

    // Epilogue: C layout -> element r of accumulator is (M = r + 8*lh, N = ll).
    v8f acc[2][2] = {{c00, c01}, {c10, c11}};
    float negd = 0.f;
    #pragma unroll
    for (int rg = 0; rg < 2; ++rg) {
        #pragma unroll
        for (int cg = 0; cg < 2; ++cg) {
            v8f cc = acc[rg][cg];
            int colg = colBase + 16 * cg + ll;
            #pragma unroll
            for (int r = 0; r < 8; ++r) {
                int rowg = rowBase + 16 * rg + (lh << 3) + r;
                float s  = cc[r];
                float nsum = 0.f;
                int   cnt  = 0;
                if ((rowg / GCLS) != (colg / GCLS)) {           // negative pair
                    negd += s;
                    if (s > min_pos[rowg] - 0.05f) {            // kept negative
                        cnt  = 1;
                        nsum = softplusf(50.f * (s - MARGIN));
                    }
                }
                // reduce across the 16 lanes sharing this row
                #pragma unroll
                for (int m = 1; m < 16; m <<= 1) {
                    nsum += __shfl_xor(nsum, m, 32);
                    cnt  += __shfl_xor(cnt,  m, 32);
                }
                if (ll == 0) {                                  // lanes 0 and 16
                    atomicAdd(&neg_sum[rowg], nsum);
                    atomicAdd(&sel_cnt[rowg], cnt);
                }
            }
        }
    }
    #pragma unroll
    for (int m = 1; m < 32; m <<= 1) negd += __shfl_xor(negd, m, 32);
    if (lane == 0) atomicAdd(neg_d_sum, negd);
}

// ---------------- final scalar reduction -----------------------------------
__global__ void finalize_kernel(const float* __restrict__ pos_loss,
                                const float* __restrict__ neg_sum,
                                const int* __restrict__ sel_cnt,
                                const float* __restrict__ scalars,
                                float* __restrict__ out, int n) {
    __shared__ float sl[256];
    __shared__ int   ss[256];
    float lsum = 0.f; int skip = 0;
    for (int i = threadIdx.x; i < n; i += 256) {
        int c = sel_cnt[i];
        if (c > 0) lsum += pos_loss[i] + 0.04f * neg_sum[i] / (float)c;
        else       skip++;
    }
    sl[threadIdx.x] = lsum; ss[threadIdx.x] = skip;
    __syncthreads();
    for (int s = 128; s > 0; s >>= 1) {
        if (threadIdx.x < s) {
            sl[threadIdx.x] += sl[threadIdx.x + s];
            ss[threadIdx.x] += ss[threadIdx.x + s];
        }
        __syncthreads();
    }
    if (threadIdx.x == 0) {
        float fn = (float)n;
        out[0] = sl[0] / fn;                                    // loss
        out[1] = (float)ss[0] / fn;                             // prec
        out[2] = scalars[0] / (fn * (float)(GCLS - 1));         // pos_d
        out[3] = scalars[1] / (fn * (float)(n - GCLS));         // neg_d
    }
}

// ---------------------------------------------------------------------------
extern "C" void kernel_launch(void* const* d_in, const int* in_sizes, int n_in,
                              void* d_out, int out_size, void* d_ws, size_t ws_size,
                              hipStream_t stream) {
    const float* x = (const float*)d_in[0];
    const int n = in_sizes[1];            // 4096
    const int d = in_sizes[0] / n;        // 1024

    // Workspace layout: bf16 hi [n*d], bf16 lo [n*d], then f32/i32 row arrays.
    char* ws = (char*)d_ws;
    __bf16* xhi    = (__bf16*)ws;
    __bf16* xlo    = (__bf16*)(ws + (size_t)n * d * sizeof(__bf16));
    float*  minpos = (float*)(ws + (size_t)n * d * 2 * sizeof(__bf16));
    float*  ploss  = minpos + n;
    float*  nsum   = ploss + n;
    int*    scnt   = (int*)(nsum + n);
    float*  scal   = (float*)(scnt + n);  // [0]=pos_d_sum, [1]=neg_d_sum

    init_kernel<<<(n + 255) / 256, 256, 0, stream>>>(nsum, scnt, scal, n);
    convert_kernel<<<(n * d + 255) / 256, 256, 0, stream>>>(x, xhi, xlo, n * d);
    minpos_kernel<<<n / 8, 256, 0, stream>>>(x, minpos, ploss, &scal[0], n, d);

    dim3 grid(n / 64, n / 128);           // (col tiles, row tiles)
    simtile_kernel<<<grid, 256, 0, stream>>>(xhi, xlo, minpos, nsum, scnt,
                                             &scal[1], n, d);

    finalize_kernel<<<1, 256, 0, stream>>>(ploss, nsum, scnt, scal,
                                           (float*)d_out, n);
}
